// GCL_71846212927514
// MI455X (gfx1250) — matrix operand order repaired and checked by reading, not compiled, not run on hardware
//
#include <hip/hip_runtime.h>
#include <hip/hip_bf16.h>

typedef _Float16 v16h __attribute__((ext_vector_type(16)));
typedef _Float16 v8h  __attribute__((ext_vector_type(8)));
typedef float    v8f  __attribute__((ext_vector_type(8)));

#define N_NODES 50000
#define N_EDGES 400000
#define DF      128
#define EDGE_TILES (N_EDGES / 16)   // 25000
#define NODE_TILES (N_NODES / 16)   // 3125

// workspace layout (bytes)
#define OFF_WE1T 0u                       // 128*256 f16 = 65536
#define OFF_WE2T 65536u                   // 128*128 f16 = 32768
#define OFF_WN1T 98304u                   // 65536
#define OFF_WN2T 163840u                  // 32768
#define OFF_AGG  196608u                  // N*128 f32 = 25,600,000
#define OFF_T1   25796608u                // E*128 f16 = 102,400,000 (reused for node 'u')

__device__ __forceinline__ float fsilu(float x) {
    return x * (1.0f / (1.0f + __expf(-x)));
}
__device__ __forceinline__ float fsigmoid(float x) {
    return 1.0f / (1.0f + __expf(-x));
}

// ---------------- weight prep: f32 [K][N] -> f16 [N][K] -----------------
__global__ void prep_weights(const float* __restrict__ We1, const float* __restrict__ We2,
                             const float* __restrict__ Wn1, const float* __restrict__ Wn2,
                             _Float16* __restrict__ We1t, _Float16* __restrict__ We2t,
                             _Float16* __restrict__ Wn1t, _Float16* __restrict__ Wn2t) {
    int t = blockIdx.x * 256 + threadIdx.x;            // 0 .. 98303
    if (t < 32768) {                                   // We1t [128][256]
        int n = t >> 8, k = t & 255;
        We1t[t] = (_Float16)We1[k * 128 + n];
    } else if (t < 49152) {                            // We2t [128][128]
        int t2 = t - 32768; int n = t2 >> 7, k = t2 & 127;
        We2t[t2] = (_Float16)We2[k * 128 + n];
    } else if (t < 81920) {                            // Wn1t [128][256]
        int t3 = t - 49152; int n = t3 >> 8, k = t3 & 255;
        Wn1t[t3] = (_Float16)Wn1[k * 128 + n];
    } else {                                           // Wn2t [128][128]
        int t4 = t - 81920; int n = t4 >> 7, k = t4 & 127;
        Wn2t[t4] = (_Float16)Wn2[k * 128 + n];
    }
}

__global__ void zero_agg(float* __restrict__ agg) {
    int t = blockIdx.x * 256 + threadIdx.x;
    if (t < N_NODES * DF) agg[t] = 0.0f;
}

// ------------- edge layer 1: t1 = silu(concat(h[row],h[col]) @ We1 + be1) -------------
__global__ __launch_bounds__(256) void edge1_kernel(
        const float* __restrict__ h, const int* __restrict__ eidx,
        const _Float16* __restrict__ We1t, const float* __restrict__ be1,
        _Float16* __restrict__ t1) {
    __shared__ _Float16 sW[128 * 256];                 // 64 KB
    int tid = threadIdx.x;
    for (int i = tid; i < 128 * 256; i += 256) sW[i] = We1t[i];
    __syncthreads();

    int lane = tid & 31, w = tid >> 5;
    int m = lane & 15, hh = lane >> 4;
    for (int tile = blockIdx.x * 8 + w; tile < EDGE_TILES; tile += gridDim.x * 8) {
        int e0 = tile * 16;
        int e  = e0 + m;
        int r  = eidx[e];
        int cc = eidx[N_EDGES + e];
        const float* pr = h + (size_t)r  * DF;
        const float* pc = h + (size_t)cc * DF;

        v16h a[8];
        #pragma unroll
        for (int kk = 0; kk < 8; ++kk) {
            const float* src = (kk < 4) ? pr : pc;
            int kb = (kk & 3) * 32 + 16 * hh;
            #pragma unroll
            for (int i = 0; i < 8; ++i) {
                a[kk][i]     = (_Float16)src[kb + i];
                a[kk][i + 8] = (_Float16)src[kb + 8 + i];
            }
        }
        #pragma unroll
        for (int nt = 0; nt < 8; ++nt) {
            int n0 = nt * 16;
            float bv = be1[n0 + m];
            v8f acc = {bv, bv, bv, bv, bv, bv, bv, bv};
            #pragma unroll
            for (int kk = 0; kk < 8; ++kk) {
                v16h b = *(const v16h*)(&sW[(n0 + m) * 256 + kk * 32 + 16 * hh]);
                acc = __builtin_amdgcn_wmma_f32_16x16x32_f16(
                        false, a[kk], false, b, (short)0, acc, false, false);
            }
            #pragma unroll
            for (int v = 0; v < 8; ++v) {
                t1[(size_t)(e0 + v + 8 * hh) * DF + n0 + m] = (_Float16)fsilu(acc[v]);
            }
        }
    }
}

// ------- edge layer 2: mij = silu(t1 @ We2 + be2); att; scatter agg -------
__global__ __launch_bounds__(256) void edge2_kernel(
        const _Float16* __restrict__ t1, const int* __restrict__ eidx,
        const float* __restrict__ emask, const _Float16* __restrict__ We2t,
        const float* __restrict__ be2, const float* __restrict__ Wa,
        const float* __restrict__ ba, float* __restrict__ agg,
        float* __restrict__ mij) {
    __shared__ _Float16 sW[128 * 128];                 // 32 KB
    __shared__ float sWa[128];
    int tid = threadIdx.x;
    for (int i = tid; i < 128 * 128; i += 256) sW[i] = We2t[i];
    if (tid < 128) sWa[tid] = Wa[tid];
    __syncthreads();
    float ba0 = ba[0];

    int lane = tid & 31, w = tid >> 5;
    int m = lane & 15, hh = lane >> 4;
    for (int tile = blockIdx.x * 8 + w; tile < EDGE_TILES; tile += gridDim.x * 8) {
        int e0 = tile * 16;
        const _Float16* pa = t1 + (size_t)(e0 + m) * DF;

        v16h a[4];
        #pragma unroll
        for (int kk = 0; kk < 4; ++kk) {
            int kb = kk * 32 + 16 * hh;
            v8h lo = *(const v8h*)(pa + kb);
            v8h hi = *(const v8h*)(pa + kb + 8);
            a[kk] = __builtin_shufflevector(lo, hi, 0, 1, 2, 3, 4, 5, 6, 7,
                                            8, 9, 10, 11, 12, 13, 14, 15);
        }
        v8f c[8];
        #pragma unroll
        for (int nt = 0; nt < 8; ++nt) {
            int n0 = nt * 16;
            float bv = be2[n0 + m];
            v8f acc = {bv, bv, bv, bv, bv, bv, bv, bv};
            #pragma unroll
            for (int kk = 0; kk < 4; ++kk) {
                v16h b = *(const v16h*)(&sW[(n0 + m) * 128 + kk * 32 + 16 * hh]);
                acc = __builtin_amdgcn_wmma_f32_16x16x32_f16(
                        false, a[kk], false, b, (short)0, acc, false, false);
            }
            #pragma unroll
            for (int v = 0; v < 8; ++v) acc[v] = fsilu(acc[v]);
            c[nt] = acc;
            #pragma unroll
            for (int v = 0; v < 8; ++v)
                mij[(size_t)(e0 + v + 8 * hh) * DF + n0 + m] = acc[v];
        }
        // attention: per-row dot(mij_row, Wa); lane holds N ≡ m (mod 16)
        float part[8];
        #pragma unroll
        for (int v = 0; v < 8; ++v) part[v] = 0.0f;
        #pragma unroll
        for (int nt = 0; nt < 8; ++nt) {
            float wv = sWa[nt * 16 + m];
            #pragma unroll
            for (int v = 0; v < 8; ++v) part[v] += c[nt][v] * wv;
        }
        #pragma unroll
        for (int off = 1; off < 16; off <<= 1) {
            #pragma unroll
            for (int v = 0; v < 8; ++v) part[v] += __shfl_xor(part[v], off, 32);
        }
        float fmul[8]; int rdst[8];
        #pragma unroll
        for (int v = 0; v < 8; ++v) {
            int e = e0 + v + 8 * hh;
            float att = fsigmoid(part[v] + ba0);
            fmul[v] = att * emask[e] * 0.01f;           // fold 1/NORM
            rdst[v] = eidx[e];                          // row (scatter target)
        }
        #pragma unroll
        for (int nt = 0; nt < 8; ++nt) {
            int n0 = nt * 16;
            #pragma unroll
            for (int v = 0; v < 8; ++v) {
                unsafeAtomicAdd(&agg[(size_t)rdst[v] * DF + n0 + m], c[nt][v] * fmul[v]);
            }
        }
    }
}

// ---------- node layer 1: u = silu(concat(h, agg) @ Wn1 + bn1) ----------
__global__ __launch_bounds__(256) void node1_kernel(
        const float* __restrict__ h, const float* __restrict__ agg,
        const _Float16* __restrict__ Wn1t, const float* __restrict__ bn1,
        _Float16* __restrict__ u) {
    __shared__ _Float16 sW[128 * 256];                 // 64 KB
    int tid = threadIdx.x;
    for (int i = tid; i < 128 * 256; i += 256) sW[i] = Wn1t[i];
    __syncthreads();

    int lane = tid & 31, w = tid >> 5;
    int m = lane & 15, hh = lane >> 4;
    for (int tile = blockIdx.x * 8 + w; tile < NODE_TILES; tile += gridDim.x * 8) {
        int r0 = tile * 16;
        const float* pr = h   + (size_t)(r0 + m) * DF;
        const float* pc = agg + (size_t)(r0 + m) * DF;

        v16h a[8];
        #pragma unroll
        for (int kk = 0; kk < 8; ++kk) {
            const float* src = (kk < 4) ? pr : pc;
            int kb = (kk & 3) * 32 + 16 * hh;
            #pragma unroll
            for (int i = 0; i < 8; ++i) {
                a[kk][i]     = (_Float16)src[kb + i];
                a[kk][i + 8] = (_Float16)src[kb + 8 + i];
            }
        }
        #pragma unroll
        for (int nt = 0; nt < 8; ++nt) {
            int n0 = nt * 16;
            float bv = bn1[n0 + m];
            v8f acc = {bv, bv, bv, bv, bv, bv, bv, bv};
            #pragma unroll
            for (int kk = 0; kk < 8; ++kk) {
                v16h b = *(const v16h*)(&sW[(n0 + m) * 256 + kk * 32 + 16 * hh]);
                acc = __builtin_amdgcn_wmma_f32_16x16x32_f16(
                        false, a[kk], false, b, (short)0, acc, false, false);
            }
            #pragma unroll
            for (int v = 0; v < 8; ++v)
                u[(size_t)(r0 + v + 8 * hh) * DF + n0 + m] = (_Float16)fsilu(acc[v]);
        }
    }
}

// ---- node layer 2: h_out = (h + u @ Wn2 + bn2) * node_mask ----
__global__ __launch_bounds__(256) void node2_kernel(
        const _Float16* __restrict__ u, const float* __restrict__ h,
        const float* __restrict__ nmask, const _Float16* __restrict__ Wn2t,
        const float* __restrict__ bn2, float* __restrict__ hout) {
    __shared__ _Float16 sW[128 * 128];                 // 32 KB
    int tid = threadIdx.x;
    for (int i = tid; i < 128 * 128; i += 256) sW[i] = Wn2t[i];
    __syncthreads();

    int lane = tid & 31, w = tid >> 5;
    int m = lane & 15, hh = lane >> 4;
    for (int tile = blockIdx.x * 8 + w; tile < NODE_TILES; tile += gridDim.x * 8) {
        int r0 = tile * 16;
        const _Float16* pa = u + (size_t)(r0 + m) * DF;

        v16h a[4];
        #pragma unroll
        for (int kk = 0; kk < 4; ++kk) {
            int kb = kk * 32 + 16 * hh;
            v8h lo = *(const v8h*)(pa + kb);
            v8h hi = *(const v8h*)(pa + kb + 8);
            a[kk] = __builtin_shufflevector(lo, hi, 0, 1, 2, 3, 4, 5, 6, 7,
                                            8, 9, 10, 11, 12, 13, 14, 15);
        }
        #pragma unroll
        for (int nt = 0; nt < 8; ++nt) {
            int n0 = nt * 16;
            float bv = bn2[n0 + m];
            v8f acc = {bv, bv, bv, bv, bv, bv, bv, bv};
            #pragma unroll
            for (int kk = 0; kk < 4; ++kk) {
                v16h b = *(const v16h*)(&sW[(n0 + m) * 128 + kk * 32 + 16 * hh]);
                acc = __builtin_amdgcn_wmma_f32_16x16x32_f16(
                        false, a[kk], false, b, (short)0, acc, false, false);
            }
            #pragma unroll
            for (int v = 0; v < 8; ++v) {
                int row = r0 + v + 8 * hh;
                float val = acc[v] + h[(size_t)row * DF + n0 + m];
                hout[(size_t)row * DF + n0 + m] = val * nmask[row];
            }
        }
    }
}

extern "C" void kernel_launch(void* const* d_in, const int* in_sizes, int n_in,
                              void* d_out, int out_size, void* d_ws, size_t ws_size,
                              hipStream_t stream) {
    (void)in_sizes; (void)n_in; (void)out_size; (void)ws_size;
    const float* h     = (const float*)d_in[0];
    const int*   eidx  = (const int*)d_in[1];
    // d_in[2] = edge_attr (0 elements, unused)
    const float* nmask = (const float*)d_in[3];
    const float* emask = (const float*)d_in[4];
    const float* We1   = (const float*)d_in[5];
    const float* be1   = (const float*)d_in[6];
    const float* We2   = (const float*)d_in[7];
    const float* be2   = (const float*)d_in[8];
    const float* Wa    = (const float*)d_in[9];
    const float* ba    = (const float*)d_in[10];
    const float* Wn1   = (const float*)d_in[11];
    const float* bn1   = (const float*)d_in[12];
    const float* Wn2   = (const float*)d_in[13];
    const float* bn2   = (const float*)d_in[14];

    char* ws = (char*)d_ws;
    _Float16* We1t = (_Float16*)(ws + OFF_WE1T);
    _Float16* We2t = (_Float16*)(ws + OFF_WE2T);
    _Float16* Wn1t = (_Float16*)(ws + OFF_WN1T);
    _Float16* Wn2t = (_Float16*)(ws + OFF_WN2T);
    float*    agg  = (float*)(ws + OFF_AGG);
    _Float16* t1   = (_Float16*)(ws + OFF_T1);      // edge hidden, reused as node 'u'

    float* hout = (float*)d_out;
    float* mij  = (float*)d_out + (size_t)N_NODES * DF;

    prep_weights<<<384, 256, 0, stream>>>(We1, We2, Wn1, Wn2, We1t, We2t, Wn1t, Wn2t);
    zero_agg<<<(N_NODES * DF) / 256, 256, 0, stream>>>(agg);
    edge1_kernel<<<EDGE_TILES / 8, 256, 0, stream>>>(h, eidx, We1t, be1, t1);
    edge2_kernel<<<EDGE_TILES / 8, 256, 0, stream>>>(t1, eidx, emask, We2t, be2, Wa, ba, agg, mij);
    node1_kernel<<<(NODE_TILES + 7) / 8, 256, 0, stream>>>(h, agg, Wn1t, bn1, t1);
    node2_kernel<<<(NODE_TILES + 7) / 8, 256, 0, stream>>>(t1, h, nmask, Wn2t, bn2, hout);
}
